// GL_multiAttention_21414706938444
// MI455X (gfx1250) — matrix operand order repaired and checked by reading, not compiled, not run on hardware
//
#include <hip/hip_runtime.h>
#include <hip/hip_bf16.h>

// Problem constants (fixed by the reference's setup_inputs()).
constexpr int B  = 4;
constexpr int F  = 256;
constexpr int U  = 2048;
constexpr int H  = 8;
constexpr int HS = 32;                 // F / H, == WMMA K dim for f16
#define MINF_C (1.0f - 4294967296.0f)  // matches torch/jax source

typedef __attribute__((ext_vector_type(16))) _Float16 v16h;
typedef __attribute__((ext_vector_type(8)))  float    v8f;

union AFrag { v16h h; unsigned u[8]; };
union Pack8 { _Float16 h[8]; uint4 v; };

// A-fragment (16x32 f16) dword offset pattern (contiguous K order):
// VGPR r holds K = 2*r + (lane>=16?16:0), +{0,1}
__device__ __forceinline__ int a_dword(int r, int hi) {
    return r + hi * 8;
}

// ---------------------------------------------------------------------------
// Stage 1: K[b,u,f] = x[b,f,u] (f16), G[b,u,f] = cummean_u x[b,f,:] (f16)
// One wave per (b,f) row; inclusive shuffle scan over U in chunks of 32.
// ---------------------------------------------------------------------------
__global__ __launch_bounds__(128) void k_scan(const float* __restrict__ x,
                                              _Float16* __restrict__ Kf,
                                              _Float16* __restrict__ Gf) {
    int wave = blockIdx.x * (blockDim.x >> 5) + (threadIdx.x >> 5);
    int lane = threadIdx.x & 31;
    if (wave >= B * F) return;
    int b = wave / F, f = wave % F;
    const float* xr = x + ((size_t)b * F + f) * U;
    float carry = 0.0f;
    for (int u0 = 0; u0 < U; u0 += 32) {
        float xv = xr[u0 + lane];
        float s = xv;
        #pragma unroll
        for (int off = 1; off < 32; off <<= 1) {
            float t = __shfl_up(s, off, 32);
            if (lane >= off) s += t;
        }
        float tot = __shfl(s, 31, 32);
        s += carry;
        carry += tot;
        float g = s / (float)(u0 + lane + 1);
        size_t o = ((size_t)b * U + u0 + lane) * F + f;
        Kf[o] = (_Float16)xv;
        Gf[o] = (_Float16)g;
    }
}

// ---------------------------------------------------------------------------
// Stage 2: Y = X @ W^T + bias.   X: [B*U, F] f16,  W: [F, F] f32 row-major.
// One wave computes one 16x16 tile; K-loop of 8 WMMAs (K=32 each).
// mode 0: store Y row-major f16 [B*U, F]   (Q path)
// mode 1: store Y transposed f16 [B, F, U] (V path, for PV B-fragments)
// ---------------------------------------------------------------------------
__global__ __launch_bounds__(128) void k_proj(const _Float16* __restrict__ X,
                                              const float* __restrict__ W,
                                              const float* __restrict__ bias,
                                              _Float16* __restrict__ Yrow,
                                              _Float16* __restrict__ Yt,
                                              int mode) {
    int wave = blockIdx.x * (blockDim.x >> 5) + (threadIdx.x >> 5);
    int lane = threadIdx.x & 31;
    int li = lane & 15, hi = lane >> 4;
    const int NT = F / 16;
    int mt = wave / NT, nt = wave % NT;
    if (mt >= (B * U) / 16) return;

    int arow = mt * 16 + li;   // global M row
    int wrow = nt * 16 + li;   // W row == output column (B-fragment N = li)
    v8f acc = {};
    for (int kb = 0; kb < F; kb += 32) {
        AFrag a;
        const unsigned* ap = (const unsigned*)(X + (size_t)arow * F + kb);
        #pragma unroll
        for (int r = 0; r < 8; r++) a.u[r] = ap[a_dword(r, hi)];
        // B-fragment: lane holds N=li, K slots e -> k = kb + hi*16 + e (contiguous)
        AFrag bf;
        const float* wp = W + (size_t)wrow * F + kb + hi * 16;
        #pragma unroll
        for (int e = 0; e < 16; e++) bf.h[e] = (_Float16)wp[e];
        acc = __builtin_amdgcn_wmma_f32_16x16x32_f16(false, a.h, false, bf.h,
                                                     (short)0, acc, false, false);
    }
    float bv = bias[wrow];
    if (mode == 0) {
        #pragma unroll
        for (int r = 0; r < 8; r++) {
            int row = mt * 16 + r + hi * 8;   // C layout: M = r + hi*8
            Yrow[(size_t)row * F + wrow] = (_Float16)(acc[r] + bv);
        }
    } else {
        // Transposed store: fixed feature column wrow, 8 consecutive u values.
        int rowg = mt * 16;
        int bidx = rowg / U;
        int u0 = (rowg % U) + hi * 8;
        Pack8 t;
        #pragma unroll
        for (int r = 0; r < 8; r++) t.h[r] = (_Float16)(acc[r] + bv);
        *(uint4*)(Yt + ((size_t)bidx * F + wrow) * U + u0) = t.v;
    }
}

// ---------------------------------------------------------------------------
// Stage 3: fused causal attention with threshold mask + online softmax.
// One wave per (b, h, 16-row tile). 32 keys per inner step:
//   2 score WMMAs (16x16x32, K = Hs) + 2 PV WMMAs (16x16x32, K = keys).
// ---------------------------------------------------------------------------
#define WPB 4
__global__ __launch_bounds__(128) void k_attn(const _Float16* __restrict__ Qf,
                                              const _Float16* __restrict__ Kf,
                                              const _Float16* __restrict__ Vt,
                                              float* __restrict__ out) {
    __shared__ _Float16 Pst[WPB][16 * 32];   // per-wave P staging (C->A relayout)
    int wave = blockIdx.x * WPB + (threadIdx.x >> 5);
    int lane = threadIdx.x & 31;
    int li = lane & 15, hi = lane >> 4;
    const int UT = U / 16;
    int ut = wave % UT;
    int h  = (wave / UT) % H;
    int b  = wave / (UT * H);
    if (b >= B) return;
    int r0 = ut * 16;

    // Q A-fragment for this 16-row tile (K dim = Hs = 32, loaded once).
    AFrag qa;
    {
        const unsigned* qp =
            (const unsigned*)(Qf + ((size_t)(b * U + r0 + li)) * F + h * HS);
        #pragma unroll
        for (int r = 0; r < 8; r++) qa.u[r] = qp[a_dword(r, hi)];
    }

    float mrun[8], lrun[8];
    v8f o0 = {}, o1 = {};
    v8f zero = {};
    #pragma unroll
    for (int r = 0; r < 8; r++) { mrun[r] = -__builtin_inff(); lrun[r] = 0.0f; }
    const float sc = 0.17677669529663688f;   // 1/sqrt(Hs)

    _Float16* P = &Pst[threadIdx.x >> 5][0];
    int jmax = (r0 + 15) / 32;
    for (int j = 0; j <= jmax; j++) {
        int v0 = j * 32;
        // K B-fragments: lane N = li -> key row v0+{0,16}+li, K slots = feature
        const v16h* kp0 =
            (const v16h*)(Kf + ((size_t)(b * U + v0 + li)) * F + h * HS + hi * 16);
        const v16h* kp1 =
            (const v16h*)(Kf + ((size_t)(b * U + v0 + 16 + li)) * F + h * HS + hi * 16);
        v16h bk0 = *kp0, bk1 = *kp1;
        if (j < jmax) {   // prefetch next K tile rows (global_prefetch_b8)
            __builtin_prefetch(Kf + ((size_t)(b * U + v0 + 32 + li)) * F + h * HS, 0, 1);
        }
        v8f s0 = __builtin_amdgcn_wmma_f32_16x16x32_f16(false, qa.h, false, bk0,
                                                        (short)0, zero, false, false);
        v8f s1 = __builtin_amdgcn_wmma_f32_16x16x32_f16(false, qa.h, false, bk1,
                                                        (short)0, zero, false, false);
        #pragma unroll
        for (int r = 0; r < 8; r++) {
            int row = r0 + r + hi * 8;        // C layout row
            float f0 = s0[r];
            f0 = (f0 <= 0.0f) ? MINF_C : f0;
            if (v0 + li > row) f0 += MINF_C;
            f0 *= sc;
            float f1 = s1[r];
            f1 = (f1 <= 0.0f) ? MINF_C : f1;
            if (v0 + 16 + li > row) f1 += MINF_C;
            f1 *= sc;
            // row lives entirely in this 16-lane half -> butterfly over 4 bits
            float m = fmaxf(f0, f1);
            #pragma unroll
            for (int off = 1; off < 16; off <<= 1) m = fmaxf(m, __shfl_xor(m, off, 32));
            float mn = fmaxf(mrun[r], m);
            float alpha = __expf(mrun[r] - mn);
            float p0 = __expf(f0 - mn);
            float p1 = __expf(f1 - mn);
            float rs = p0 + p1;
            #pragma unroll
            for (int off = 1; off < 16; off <<= 1) rs += __shfl_xor(rs, off, 32);
            lrun[r] = lrun[r] * alpha + rs;
            mrun[r] = mn;
            o0[r] *= alpha;
            o1[r] *= alpha;
            P[(r + hi * 8) * 32 + li]      = (_Float16)p0;
            P[(r + hi * 8) * 32 + 16 + li] = (_Float16)p1;
        }
        // Same-wave LDS: DS ops are in-order, but fence the compiler + counters.
        asm volatile("s_wait_dscnt 0" ::: "memory");
        // P A-fragment (K dim = 32 key slots)
        AFrag pf;
        const unsigned* pp = (const unsigned*)(P + li * 32);
        #pragma unroll
        for (int r = 0; r < 8; r++) pf.u[r] = pp[a_dword(r, hi)];
        // V B-fragments from transposed V: lane N = feature, K slots = key idx
        const v16h* vp0 =
            (const v16h*)(Vt + ((size_t)(b * F) + h * HS + li) * U + v0 + hi * 16);
        const v16h* vp1 =
            (const v16h*)(Vt + ((size_t)(b * F) + h * HS + 16 + li) * U + v0 + hi * 16);
        v16h bv0 = *vp0, bv1 = *vp1;
        o0 = __builtin_amdgcn_wmma_f32_16x16x32_f16(false, pf.h, false, bv0,
                                                    (short)0, o0, false, false);
        o1 = __builtin_amdgcn_wmma_f32_16x16x32_f16(false, pf.h, false, bv1,
                                                    (short)0, o1, false, false);
    }
    // Final normalization + fp32 store: out[b, u, h*HS + {li, 16+li}]
    #pragma unroll
    for (int r = 0; r < 8; r++) {
        int u = r0 + r + hi * 8;
        float inv = 1.0f / lrun[r];
        float* op = out + ((size_t)(b * U + u)) * F + h * HS;
        op[li]      = o0[r] * inv;
        op[16 + li] = o1[r] * inv;
    }
}

// ---------------------------------------------------------------------------
extern "C" void kernel_launch(void* const* d_in, const int* in_sizes, int n_in,
                              void* d_out, int out_size, void* d_ws, size_t ws_size,
                              hipStream_t stream) {
    (void)in_sizes; (void)n_in; (void)out_size; (void)ws_size;
    const float* x    = (const float*)d_in[0];
    const float* Wq_w = (const float*)d_in[1];
    const float* Wq_b = (const float*)d_in[2];
    const float* Wv_w = (const float*)d_in[3];
    const float* Wv_b = (const float*)d_in[4];

    // Workspace layout (f16): G, K row-major [B,U,F]; Q row-major; V^T [B,F,U].
    // Total = 4 * B*U*F * 2B = 16 MiB.
    char* ws = (char*)d_ws;
    const size_t NE = (size_t)B * U * F;
    _Float16* Gf = (_Float16*)(ws);
    _Float16* Kf = (_Float16*)(ws + NE * 2);
    _Float16* Qf = (_Float16*)(ws + NE * 4);
    _Float16* Vt = (_Float16*)(ws + NE * 6);

    k_scan<<<(B * F) / 4, 128, 0, stream>>>(x, Kf, Gf);

    int tiles = ((B * U) / 16) * (F / 16);
    k_proj<<<tiles / 4, 128, 0, stream>>>(Gf, Wq_w, Wq_b, Qf, nullptr, 0);
    k_proj<<<tiles / 4, 128, 0, stream>>>(Kf, Wv_w, Wv_b, nullptr, Vt, 1);

    int awaves = B * H * (U / 16);
    k_attn<<<awaves / WPB, 128, 0, stream>>>(Qf, Kf, Vt, (float*)d_out);
}
